// OptimizedTransformerBlock_43490838839469
// MI455X (gfx1250) — matrix operand order repaired and checked by reading, not compiled, not run on hardware
//
#include <hip/hip_runtime.h>
#include <hip/hip_bf16.h>

typedef __attribute__((ext_vector_type(16))) _Float16 v16h;
typedef __attribute__((ext_vector_type(8)))  _Float16 v8h;
typedef __attribute__((ext_vector_type(8)))  float    v8f;

#define E_DIM 1024
#define S_LEN 2048
#define NHEAD 16
#define HDIM  64
#define FF_DIM 4096
#define BATCH 2
#define MTOK  (BATCH * S_LEN)   // 4096 tokens

// ---------------------------------------------------------------------------
// WMMA helpers (gfx1250, wave32): D = A(16x32 f16) * B(32x16 f16) + C(16x16 f32)
// ---------------------------------------------------------------------------
__device__ __forceinline__ v8f wmma16(v16h a, v16h b, v8f c) {
  return __builtin_amdgcn_wmma_f32_16x16x32_f16(false, a, false, b, (short)0, c,
                                                false, false);
}

// A fragment: lane L holds row M=L%16; K = {g*8..g*8+7} and {16+g*8..16+g*8+7},
// g = L/16  ->  two contiguous 16B loads from a row-major [M,K] matrix.
__device__ __forceinline__ v16h load_a_frag(const _Float16* p, int ld, int lane) {
  int row = lane & 15, g = lane >> 4;
  const _Float16* r = p + (size_t)row * ld + g * 8;
  v8h lo = *(const v8h*)r;
  v8h hi = *(const v8h*)(r + 16);
  return __builtin_shufflevector(lo, hi, 0,1,2,3,4,5,6,7,8,9,10,11,12,13,14,15);
}

// B fragment: lane L holds column N=L%16; K = {g*8..g*8+7} and {16+g*8..16+g*8+7}.
// For a weight stored [N,K] row-major this is two contiguous 16B loads per lane.
__device__ __forceinline__ v16h load_b_frag(const _Float16* p, int ld, int lane) {
  int n = lane & 15, g = lane >> 4;
  const _Float16* r = p + (size_t)n * ld + g * 8;
  v8h lo = *(const v8h*)r;
  v8h hi = *(const v8h*)(r + 16);
  return __builtin_shufflevector(lo, hi, 0,1,2,3,4,5,6,7,8,9,10,11,12,13,14,15);
}

// ---------------------------------------------------------------------------
// f32 -> f16 conversion (weights / staging)
// ---------------------------------------------------------------------------
__global__ void cvt_kernel(const float* __restrict__ in, _Float16* __restrict__ out,
                           int n) {
  int i = blockIdx.x * blockDim.x + threadIdx.x;
  int stride = gridDim.x * blockDim.x;
  for (; i < n; i += stride) out[i] = (_Float16)in[i];
}

// ---------------------------------------------------------------------------
// LayerNorm: one block (256 threads) per row of 1024; f32 in, f16 out
// ---------------------------------------------------------------------------
__global__ __launch_bounds__(256) void ln_kernel(const float* __restrict__ x,
                                                 const float* __restrict__ sc,
                                                 const float* __restrict__ bi,
                                                 _Float16* __restrict__ out) {
  __shared__ float red[8];
  int row = blockIdx.x;
  const float* xr = x + (size_t)row * E_DIM;
  float v[4];
  float sum = 0.f;
#pragma unroll
  for (int i = 0; i < 4; i++) {
    v[i] = xr[threadIdx.x + i * 256];
    sum += v[i];
  }
#pragma unroll
  for (int m = 16; m >= 1; m >>= 1) sum += __shfl_xor(sum, m);
  int wave = threadIdx.x >> 5, lane = threadIdx.x & 31;
  if (lane == 0) red[wave] = sum;
  __syncthreads();
  float tot = red[0]+red[1]+red[2]+red[3]+red[4]+red[5]+red[6]+red[7];
  float mu = tot * (1.f / E_DIM);
  float var = 0.f;
#pragma unroll
  for (int i = 0; i < 4; i++) { float d = v[i] - mu; var += d * d; }
#pragma unroll
  for (int m = 16; m >= 1; m >>= 1) var += __shfl_xor(var, m);
  __syncthreads();
  if (lane == 0) red[wave] = var;
  __syncthreads();
  float vtot = red[0]+red[1]+red[2]+red[3]+red[4]+red[5]+red[6]+red[7];
  float inv = rsqrtf(vtot * (1.f / E_DIM) + 1e-5f);
#pragma unroll
  for (int i = 0; i < 4; i++) {
    int c = threadIdx.x + i * 256;
    out[(size_t)row * E_DIM + c] = (_Float16)((v[i] - mu) * inv * sc[c] + bi[c]);
  }
}

// ---------------------------------------------------------------------------
// Generic WMMA GEMM: C[M,N] = A[M,K](f16) @ W[N,K](f16)^T
// block = 128 threads (4 waves), tile 128x128, 64x64 per wave, K-step 32.
// EPI: 0 = QKV scatter (f16 -> Q,K [B,H,S,D], V^T [B,H,D,S])
//      1 = residual add, f32 out
//      2 = exact GELU, f16 out
// ---------------------------------------------------------------------------
template <int EPI>
__global__ __launch_bounds__(128) __attribute__((amdgpu_waves_per_eu(1)))
void gemm_wmma(const _Float16* __restrict__ A, const _Float16* __restrict__ W,
               int M, int N, int K,
               const float* __restrict__ res, float* __restrict__ outF,
               _Float16* __restrict__ outH,
               _Float16* __restrict__ qo, _Float16* __restrict__ ko,
               _Float16* __restrict__ vTo) {
  int lane = threadIdx.x & 31;
  int wave = threadIdx.x >> 5;
  int m0 = blockIdx.y * 128 + (wave >> 1) * 64;
  int n0 = blockIdx.x * 128 + (wave & 1) * 64;
  int col = lane & 15, gl = lane >> 4;

  const _Float16* Ab = A + (size_t)m0 * K;
  const _Float16* Wb = W + (size_t)n0 * K;

  v8f acc[4][4] = {};
  for (int k0 = 0; k0 < K; k0 += 32) {
    if (k0 + 64 < K) {
      __builtin_prefetch(Ab + k0 + 64, 0, 1);
      __builtin_prefetch(Wb + k0 + 64, 0, 1);
    }
    v16h a[4], b[4];
#pragma unroll
    for (int i = 0; i < 4; i++)
      a[i] = load_a_frag(Ab + (size_t)i * 16 * K + k0, K, lane);
#pragma unroll
    for (int i = 0; i < 4; i++)
      b[i] = load_b_frag(Wb + (size_t)i * 16 * K + k0, K, lane);
#pragma unroll
    for (int mi = 0; mi < 4; mi++)
#pragma unroll
      for (int ni = 0; ni < 4; ni++)
        acc[mi][ni] = wmma16(a[mi], b[ni], acc[mi][ni]);
  }

  // Epilogue. C layout: VGPR r -> M = r (lanes 0-15) / r+8 (lanes 16-31), N = lane%16.
  // A 128-row tile never crosses a batch boundary (S_LEN % 128 == 0).
  int bq = m0 >> 11;                 // batch index, constant per block
  int sb = (m0 & (S_LEN - 1)) + 8 * gl;
#pragma unroll
  for (int mi = 0; mi < 4; mi++) {
#pragma unroll
    for (int ni = 0; ni < 4; ni++) {
      int n = n0 + ni * 16 + col;
      if constexpr (EPI == 0) {
        int part = n >> 10;          // / E_DIM : 0=Q, 1=K, 2=V
        int e = n & (E_DIM - 1);
        int hh = e >> 6;             // head
        int d = e & (HDIM - 1);
        int s0r = sb + mi * 16;
        if (part == 2) {
          size_t vbase = (((size_t)(bq * NHEAD + hh) * HDIM) + d) * S_LEN + s0r;
#pragma unroll
          for (int r = 0; r < 8; r++)
            vTo[vbase + r] = (_Float16)acc[mi][ni][r];
        } else {
          _Float16* dst = (part == 0) ? qo : ko;
          size_t qbase = (((size_t)(bq * NHEAD + hh) * S_LEN) + s0r) * HDIM + d;
#pragma unroll
          for (int r = 0; r < 8; r++)
            dst[qbase + (size_t)r * HDIM] = (_Float16)acc[mi][ni][r];
        }
      } else if constexpr (EPI == 1) {
        size_t idx0 = (size_t)(m0 + mi * 16 + 8 * gl) * N + n;
#pragma unroll
        for (int r = 0; r < 8; r++) {
          size_t idx = idx0 + (size_t)r * N;
          outF[idx] = res[idx] + acc[mi][ni][r];
        }
      } else {
        size_t idx0 = (size_t)(m0 + mi * 16 + 8 * gl) * N + n;
#pragma unroll
        for (int r = 0; r < 8; r++) {
          float v = acc[mi][ni][r];
          float g = 0.5f * v * (1.0f + erff(v * 0.70710678118654752f));
          outH[idx0 + (size_t)r * N] = (_Float16)g;
        }
      }
    }
  }
}

// ---------------------------------------------------------------------------
// Flash attention: one wave per 16-query tile; online softmax, 64 keys/step.
// Q,K: [B,H,S,D] f16;  Vt: [B,H,D,S] f16;  out: attn [B*S, E] f16.
// 16 WMMAs per 64-key step; stats reductions amortized over 64 columns.
// ---------------------------------------------------------------------------
__global__ __launch_bounds__(128) void attn_kernel(const _Float16* __restrict__ Q,
                                                   const _Float16* __restrict__ Km,
                                                   const _Float16* __restrict__ Vt,
                                                   _Float16* __restrict__ attn) {
  __shared__ _Float16 plds[4 * 16 * 64];   // per-wave 16x64 P tile
  int lane = threadIdx.x & 31;
  int wave = threadIdx.x >> 5;
  int col = lane & 15, gl = lane >> 4;

  int tile = blockIdx.x * 4 + wave;   // 0..4095
  int bh = tile >> 7;                 // batch*head
  int qt = tile & 127;                // query tile within sequence
  int bq = bh >> 4;
  int hh = bh & 15;

  const _Float16* Qb = Q + ((size_t)bh * S_LEN + qt * 16) * HDIM;
  const _Float16* Kb = Km + (size_t)bh * S_LEN * HDIM;
  const _Float16* Vb = Vt + (size_t)bh * HDIM * S_LEN;
  _Float16* pl = plds + wave * 16 * 64;

  v16h aq0 = load_a_frag(Qb, HDIM, lane);       // d 0..31
  v16h aq1 = load_a_frag(Qb + 32, HDIM, lane);  // d 32..63

  v8f o0 = {}, o1 = {}, o2 = {}, o3 = {};
  float mr[8], lr[8];
#pragma unroll
  for (int r = 0; r < 8; r++) { mr[r] = -1e30f; lr[r] = 0.f; }

  for (int j = 0; j < S_LEN; j += 64) {
    // scores for keys j..j+63: 4 C tiles, dot over D=64 (2 chained WMMAs each)
    v8f s[4];
#pragma unroll
    for (int t = 0; t < 4; t++) {
      v8f z = {};
      z = wmma16(aq0, load_b_frag(Kb + (size_t)(j + 16 * t) * HDIM, HDIM, lane), z);
      z = wmma16(aq1, load_b_frag(Kb + (size_t)(j + 16 * t) * HDIM + 32, HDIM, lane), z);
      s[t] = z;
    }

    // online softmax update, one reduction pass per 64 keys
#pragma unroll
    for (int r = 0; r < 8; r++) {
      float sv[4];
#pragma unroll
      for (int t = 0; t < 4; t++) sv[t] = s[t][r] * 0.125f;   // 1/sqrt(64)
      float tm = fmaxf(fmaxf(sv[0], sv[1]), fmaxf(sv[2], sv[3]));
#pragma unroll
      for (int ms = 8; ms >= 1; ms >>= 1) tm = fmaxf(tm, __shfl_xor(tm, ms));
      float mn = fmaxf(mr[r], tm);
      float scl = __expf(mr[r] - mn);
      mr[r] = mn;
      float p[4];
#pragma unroll
      for (int t = 0; t < 4; t++) p[t] = __expf(sv[t] - mn);
      float su = (p[0] + p[1]) + (p[2] + p[3]);
#pragma unroll
      for (int ms = 8; ms >= 1; ms >>= 1) su += __shfl_xor(su, ms);
      lr[r] = lr[r] * scl + su;
      o0[r] *= scl; o1[r] *= scl; o2[r] *= scl; o3[r] *= scl;
      int mrow = r + 8 * gl;
#pragma unroll
      for (int t = 0; t < 4; t++)
        pl[mrow * 64 + t * 16 + col] = (_Float16)p[t];
    }
    asm volatile("s_wait_dscnt 0" ::: "memory");

    // reload P (16x64) in A-fragment layout from LDS: two 32-key dot chunks
    const _Float16* pr = pl + (size_t)col * 64 + gl * 8;
    v8h lo0 = *(const v8h*)pr;
    v8h hi0 = *(const v8h*)(pr + 16);
    v16h ap0 = __builtin_shufflevector(lo0, hi0,
        0,1,2,3,4,5,6,7,8,9,10,11,12,13,14,15);
    v8h lo1 = *(const v8h*)(pr + 32);
    v8h hi1 = *(const v8h*)(pr + 48);
    v16h ap1 = __builtin_shufflevector(lo1, hi1,
        0,1,2,3,4,5,6,7,8,9,10,11,12,13,14,15);

    // O += P @ V  (V^T layout makes B-frag loads contiguous in s)
    o0 = wmma16(ap0, load_b_frag(Vb + (size_t)0 * 16 * S_LEN + j, S_LEN, lane), o0);
    o1 = wmma16(ap0, load_b_frag(Vb + (size_t)1 * 16 * S_LEN + j, S_LEN, lane), o1);
    o2 = wmma16(ap0, load_b_frag(Vb + (size_t)2 * 16 * S_LEN + j, S_LEN, lane), o2);
    o3 = wmma16(ap0, load_b_frag(Vb + (size_t)3 * 16 * S_LEN + j, S_LEN, lane), o3);
    o0 = wmma16(ap1, load_b_frag(Vb + (size_t)0 * 16 * S_LEN + j + 32, S_LEN, lane), o0);
    o1 = wmma16(ap1, load_b_frag(Vb + (size_t)1 * 16 * S_LEN + j + 32, S_LEN, lane), o1);
    o2 = wmma16(ap1, load_b_frag(Vb + (size_t)2 * 16 * S_LEN + j + 32, S_LEN, lane), o2);
    o3 = wmma16(ap1, load_b_frag(Vb + (size_t)3 * 16 * S_LEN + j + 32, S_LEN, lane), o3);
  }

  // normalize and write to attn [B*S, E] (already the (0,2,1,3) transpose)
#pragma unroll
  for (int r = 0; r < 8; r++) {
    float inv = 1.f / lr[r];
    int m = qt * 16 + r + 8 * gl;
    size_t base = ((size_t)bq * S_LEN + m) * E_DIM + hh * HDIM;
    attn[base + 0  + col] = (_Float16)(o0[r] * inv);
    attn[base + 16 + col] = (_Float16)(o1[r] * inv);
    attn[base + 32 + col] = (_Float16)(o2[r] * inv);
    attn[base + 48 + col] = (_Float16)(o3[r] * inv);
  }
}

// ---------------------------------------------------------------------------
// Host-side orchestration
// ---------------------------------------------------------------------------
extern "C" void kernel_launch(void* const* d_in, const int* in_sizes, int n_in,
                              void* d_out, int out_size, void* d_ws, size_t ws_size,
                              hipStream_t stream) {
  const float* x      = (const float*)d_in[0];
  const float* qkv_w  = (const float*)d_in[1];
  const float* out_w  = (const float*)d_in[2];
  const float* up_w   = (const float*)d_in[3];
  const float* down_w = (const float*)d_in[4];
  const float* n1s    = (const float*)d_in[5];
  const float* n1b    = (const float*)d_in[6];
  const float* n2s    = (const float*)d_in[7];
  const float* n2b    = (const float*)d_in[8];
  float* out = (float*)d_out;

  char* ws = (char*)d_ws;
  size_t off = 0;
  auto alloc = [&](size_t bytes) -> char* {
    char* p = ws + off;
    off += (bytes + 255) & ~(size_t)255;
    return p;
  };
  const size_t NE = (size_t)MTOK * E_DIM;           // 4096*1024
  _Float16* qkv_h  = (_Float16*)alloc((size_t)3 * E_DIM * E_DIM * 2);
  _Float16* out_h  = (_Float16*)alloc((size_t)E_DIM * E_DIM * 2);
  _Float16* up_h   = (_Float16*)alloc((size_t)FF_DIM * E_DIM * 2);
  _Float16* down_h = (_Float16*)alloc((size_t)E_DIM * FF_DIM * 2);
  _Float16* h1_h   = (_Float16*)alloc(NE * 2);
  _Float16* q_h    = (_Float16*)alloc(NE * 2);
  _Float16* k_h    = (_Float16*)alloc(NE * 2);
  _Float16* vT_h   = (_Float16*)alloc(NE * 2);
  _Float16* attn_h = (_Float16*)alloc(NE * 2);
  float*    x1_f   = (float*)   alloc(NE * 4);
  _Float16* h2_h   = (_Float16*)alloc(NE * 2);
  _Float16* g_h    = (_Float16*)alloc((size_t)MTOK * FF_DIM * 2);
  (void)ws_size; (void)n_in; (void)in_sizes; (void)out_size;

  // 1) weights -> f16
  cvt_kernel<<<8192, 256, 0, stream>>>(qkv_w, qkv_h, 3 * E_DIM * E_DIM);
  cvt_kernel<<<4096, 256, 0, stream>>>(out_w, out_h, E_DIM * E_DIM);
  cvt_kernel<<<8192, 256, 0, stream>>>(up_w, up_h, FF_DIM * E_DIM);
  cvt_kernel<<<8192, 256, 0, stream>>>(down_w, down_h, E_DIM * FF_DIM);

  // 2) LN1
  ln_kernel<<<MTOK, 256, 0, stream>>>(x, n1s, n1b, h1_h);

  // 3) QKV projection (scatter epilogue)
  gemm_wmma<0><<<dim3(3 * E_DIM / 128, MTOK / 128), 128, 0, stream>>>(
      h1_h, qkv_h, MTOK, 3 * E_DIM, E_DIM,
      nullptr, nullptr, nullptr, q_h, k_h, vT_h);

  // 4) attention
  attn_kernel<<<MTOK * NHEAD / (16 * 4), 128, 0, stream>>>(q_h, k_h, vT_h, attn_h);

  // 5) out projection + residual(x) -> x1 (f32)
  gemm_wmma<1><<<dim3(E_DIM / 128, MTOK / 128), 128, 0, stream>>>(
      attn_h, out_h, MTOK, E_DIM, E_DIM,
      x, x1_f, nullptr, nullptr, nullptr, nullptr);

  // 6) LN2
  ln_kernel<<<MTOK, 256, 0, stream>>>(x1_f, n2s, n2b, h2_h);

  // 7) FFN up + GELU -> f16
  gemm_wmma<2><<<dim3(FF_DIM / 128, MTOK / 128), 128, 0, stream>>>(
      h2_h, up_h, MTOK, FF_DIM, E_DIM,
      nullptr, nullptr, g_h, nullptr, nullptr, nullptr);

  // 8) FFN down + residual(x1) -> out (f32)
  gemm_wmma<1><<<dim3(E_DIM / 128, MTOK / 128), 128, 0, stream>>>(
      g_h, down_h, MTOK, E_DIM, FF_DIM,
      x1_f, out, nullptr, nullptr, nullptr, nullptr);
}
